// DecoderMultiHeadAttention_32366873543153
// MI455X (gfx1250) — matrix operand order repaired and checked by reading, not compiled, not run on hardware
//
#include <hip/hip_runtime.h>

#define BB 2
#define SS 2048
#define EE 1024
#define HH 16
#define DD 64
#define OO 64
#define HO (HH*OO)

typedef __bf16 bf16_t;
typedef __attribute__((ext_vector_type(16))) __bf16 v16bf;
typedef __attribute__((ext_vector_type(8)))  __bf16 v8bf;
typedef __attribute__((ext_vector_type(8)))  float  v8f;

union V16U8 { v16bf v; unsigned u[8]; };

static __device__ inline v8f wmma_bf16(v16bf a, v16bf b, v8f c) {
  // 8 args: (neg_a, A, neg_b, B, c_mod, C, reuse_a, reuse_b)
  return __builtin_amdgcn_wmma_f32_16x16x32_bf16(false, a, false, b, (short)0, c, false, false);
}

static __device__ inline v16bf pack16(v8bf lo, v8bf hi) {
  v16bf r;
#pragma unroll
  for (int i = 0; i < 8; ++i) { r[i] = lo[i]; r[8 + i] = hi[i]; }
  return r;
}

static __device__ inline unsigned pack2(float a, float b) {
  union { __bf16 h[2]; unsigned u; } t;
  t.h[0] = (__bf16)a; t.h[1] = (__bf16)b;
  return t.u;
}

// Low 32 bits of a generic LDS pointer = LDS byte offset (ISA aperture mapping).
static __device__ inline unsigned lds_off(const void* p) {
  return (unsigned)(unsigned long long)p;
}

// Async global->LDS 16-byte copy (VGLOBAL GLOBAL_LOAD_ASYNC_TO_LDS_B128, ASYNCcnt).
static __device__ inline void async_b128(unsigned lds_byte_off, const void* gaddr) {
  asm volatile("global_load_async_to_lds_b128 %0, %1, off"
               :: "v"(lds_byte_off), "v"(gaddr)
               : "memory");
}

static __device__ inline void wait_async0() {
  asm volatile("s_wait_asynccnt 0x0" ::: "memory");
}

// A fragment: 16(M)x32(K) bf16 from row-major [M][K], leading dim ld.
// k order used here (contiguous, same as load_bt): lanes 0-15 hold K 0..15, lanes 16-31 hold K 16..31.
static __device__ inline v16bf load_a(const bf16_t* base, int ld, int lane) {
  const int row = lane & 15;
  const int kb  = (lane < 16) ? 0 : 16;
  const bf16_t* p = base + (long)row * ld + kb;
  v8bf lo = *reinterpret_cast<const v8bf*>(p);
  v8bf hi = *reinterpret_cast<const v8bf*>(p + 8);
  return pack16(lo, hi);
}

// B fragment: 32(K)x16(N) bf16 read from [N][K] storage (B transposed), leading dim ld.
// k order used here (contiguous, same as load_a): lane=column, lanes 0-15 hold K 0..15, lanes 16-31 hold K 16..31.
static __device__ inline v16bf load_bt(const bf16_t* base, int ld, int lane) {
  const int col = lane & 15;
  const int kb  = (lane < 16) ? 0 : 16;
  const bf16_t* p = base + (long)col * ld + kb;
  v8bf lo = *reinterpret_cast<const v8bf*>(p);
  v8bf hi = *reinterpret_cast<const v8bf*>(p + 8);
  return pack16(lo, hi);
}

// Online-softmax over one 32-key chunk for one 16-query tile (transposed layout:
// lane = query column, register elements = keys). Updates per-lane m/l, returns
// the P^T B-fragment (built via packed half-wave exchanges) and the rescale factor.
static __device__ inline v16bf softmax_bfrag(const v8f& c0, const v8f& c1,
                                             const float* km0, const float* km1,
                                             int lane, float& m, float& l, float& corr) {
  float s0e[8], s1e[8];
  float mx = -1e30f;
#pragma unroll
  for (int i = 0; i < 8; ++i) {
    s0e[i] = (km0[i] != 0.f) ? c0[i] * 0.125f : -1e30f;   // 1/sqrt(D=64)
    s1e[i] = (km1[i] != 0.f) ? c1[i] * 0.125f : -1e30f;
    mx = fmaxf(mx, fmaxf(s0e[i], s1e[i]));
  }
  mx = fmaxf(mx, __shfl_xor(mx, 16));
  float mn = fmaxf(m, mx);
  corr = __expf(m - mn);
  m = mn;

  float p0[8], p1[8];
  float rs = 0.f;
#pragma unroll
  for (int i = 0; i < 8; ++i) {
    p0[i] = (km0[i] != 0.f) ? __expf(s0e[i] - mn) : 0.f;
    p1[i] = (km1[i] != 0.f) ? __expf(s1e[i] - mn) : 0.f;
    rs += p0[i] + p1[i];
  }
  rs += __shfl_xor(rs, 16);
  l = l * corr + rs;

  // lanes<16 need K(t)=0..15  = {own c0, c0 from lane+16}
  // lanes>=16 need K(t)=16..31 = {c1 from lane-16, own c1}
  unsigned pc0[4], pc1[4];
#pragma unroll
  for (int j = 0; j < 4; ++j) {
    pc0[j] = pack2(p0[2 * j], p0[2 * j + 1]);
    pc1[j] = pack2(p1[2 * j], p1[2 * j + 1]);
  }
  V16U8 bp;
#pragma unroll
  for (int j = 0; j < 4; ++j) {
    unsigned x0 = (unsigned)__shfl_xor((int)pc0[j], 16);
    unsigned x1 = (unsigned)__shfl_xor((int)pc1[j], 16);
    bp.u[j]     = (lane < 16) ? pc0[j] : x1;
    bp.u[4 + j] = (lane < 16) ? x0 : pc1[j];
  }
  return bp.v;
}

// ---------------- prep kernels ----------------

__global__ void cvt_f32_bf16(const float* __restrict__ src, bf16_t* __restrict__ dst, int n) {
  int i = blockIdx.x * blockDim.x + threadIdx.x;
  if (i < n) dst[i] = (bf16_t)src[i];
}

// src fp32 [P][R][C] -> dst bf16 [P][C][R]   (P = gridDim.z planes)
__global__ void transpose_cvt(const float* __restrict__ src, bf16_t* __restrict__ dst,
                              int R, int C) {
  long base = (long)blockIdx.z * R * C;
  int i = blockIdx.x * blockDim.x + threadIdx.x;
  if (i < R * C) {
    int r = i / C, c = i % C;
    dst[base + (long)c * R + r] = (bf16_t)src[base + i];
  }
}

// ---------------- QKV projection: [B*S,E] x [E,D] per head, 32 rows/wave ----------------

__global__ __launch_bounds__(32) void qkv_proj(
    const bf16_t* __restrict__ qx, const bf16_t* __restrict__ kx, const bf16_t* __restrict__ vx,
    const bf16_t* __restrict__ WqT, const bf16_t* __restrict__ WkT, const bf16_t* __restrict__ WvT,
    const float* __restrict__ bq, const float* __restrict__ bk, const float* __restrict__ bv,
    bf16_t* __restrict__ qh, bf16_t* __restrict__ kh, bf16_t* __restrict__ vhT)
{
  const int lane  = threadIdx.x;
  const int m0    = blockIdx.x * 32;   // global row in [0, B*S)
  const int h     = blockIdx.y;
  const int which = blockIdx.z;        // 0=q 1=k 2=v

  const bf16_t* act  = (which == 0) ? qx : (which == 1) ? kx : vx;
  const bf16_t* W    = ((which == 0) ? WqT : (which == 1) ? WkT : WvT) + (long)h * DD * EE;
  const float*  bias = ((which == 0) ? bq : (which == 1) ? bk : bv) + h * DD;

  v8f acc[2][4] = {};
  for (int k0 = 0; k0 < EE; k0 += 32) {
    v16bf a0 = load_a(act + (long)m0 * EE + k0, EE, lane);
    v16bf a1 = load_a(act + (long)(m0 + 16) * EE + k0, EE, lane);
#pragma unroll
    for (int n = 0; n < 4; ++n) {
      v16bf b = load_bt(W + (long)(n * 16) * EE + k0, EE, lane);   // shared by both M tiles
      acc[0][n] = wmma_bf16(a0, b, acc[0][n]);
      acc[1][n] = wmma_bf16(a1, b, acc[1][n]);
    }
  }
  const int half8 = (lane < 16) ? 0 : 8;
  const int col   = lane & 15;
#pragma unroll
  for (int mt = 0; mt < 2; ++mt) {
#pragma unroll
    for (int n = 0; n < 4; ++n) {
      int d = n * 16 + col;
      float bi = bias[d];
#pragma unroll
      for (int i = 0; i < 8; ++i) {
        int g  = m0 + mt * 16 + i + half8;
        int b_ = g / SS, s = g % SS;
        float val = acc[mt][n][i] + bi;
        if (which == 2) {
          vhT[(((long)b_ * HH + h) * DD + d) * SS + s] = (bf16_t)val;  // v transposed [d][t]
        } else {
          bf16_t* dst = (which == 0) ? qh : kh;
          dst[(((long)b_ * HH + h) * SS + s) * DD + d] = (bf16_t)val;  // [s][d]
        }
      }
    }
  }
}

// ---------------- flash attention: 4 waves/WG, async-staged K/V in LDS ----------------
// S^T[t][s] = K x Q^T (softmax dim in registers); o^T[d][s] = V^T x P^T.
// Each workgroup covers 128 query columns; K/V chunks staged once per WG via
// GLOBAL_LOAD_ASYNC_TO_LDS_B128 (double-buffered, overlapped with WMMA compute).

__global__ __launch_bounds__(128) void attn(
    const bf16_t* __restrict__ qh, const bf16_t* __restrict__ kh, const bf16_t* __restrict__ vhT,
    const float* __restrict__ qmask, const float* __restrict__ kmask,
    bf16_t* __restrict__ ob)
{
  __shared__ __align__(16) bf16_t skb[2][32 * DD];  // K chunk [t][d], 4KB each
  __shared__ __align__(16) bf16_t svb[2][DD * 32];  // V chunk [d][t], 4KB each

  const int tid   = threadIdx.x;
  const int lane  = tid & 31;
  const int wave  = tid >> 5;
  const int s0    = blockIdx.x * 128 + wave * 32;
  const int h     = blockIdx.y;
  const int b     = blockIdx.z;
  const int half8 = (lane < 16) ? 0 : 8;
  const int col   = lane & 15;

  const bf16_t* qbase = qh  + (((long)b * HH + h) * SS + s0) * DD;
  const bf16_t* kbase = kh  + (((long)b * HH + h) * SS) * DD;
  const bf16_t* vbase = vhT + (((long)b * HH + h) * DD) * SS;

  // Q as B-fragments (lane = query column, K = d), loaded once; two query tiles/wave
  v16bf bq0a = load_bt(qbase, DD, lane);
  v16bf bq0b = load_bt(qbase + 32, DD, lane);
  v16bf bq1a = load_bt(qbase + 16 * DD, DD, lane);
  v16bf bq1b = load_bt(qbase + 16 * DD + 32, DD, lane);

  // cooperative async staging: 128 threads x (2+2) 16B units per chunk
  auto stage = [&](int t0, int buf) {
    const char* kg = (const char*)(kbase + (long)t0 * DD);   // 4KB contiguous
    unsigned kl = lds_off(&skb[buf][0]);
#pragma unroll
    for (int r = 0; r < 2; ++r) {
      int u = tid + r * 128;                                  // 0..255
      async_b128(kl + u * 16, kg + u * 16);
    }
    unsigned vl = lds_off(&svb[buf][0]);
#pragma unroll
    for (int r = 0; r < 2; ++r) {
      int u  = tid + r * 128;                                 // 0..255
      int dr = u >> 2;                                        // d row 0..63
      int qd = u & 3;                                         // 16B quarter of the row
      const char* vg = (const char*)(vbase + (long)dr * SS + t0) + qd * 16;
      async_b128(vl + (unsigned)(dr * 32 + qd * 8) * 2, vg);
    }
  };

  float m0v = -1e30f, l0v = 0.f;   // per-lane scalars, query tile 0
  float m1v = -1e30f, l1v = 0.f;   // query tile 1
  v8f acc0[4] = {}, acc1[4] = {};

  stage(0, 0);
  const int nch = SS / 32;
  for (int ci = 0; ci < nch; ++ci) {
    const int t0  = ci * 32;
    const int buf = ci & 1;
    wait_async0();          // this wave's copies for buf have landed in LDS
    __syncthreads();        // ... and everyone else's
    if (ci + 1 < nch) stage(t0 + 32, buf ^ 1);   // overlap next DMA with compute

    const bf16_t* kc = &skb[buf][0];
    const bf16_t* vc = &svb[buf][0];

    // ---- K fragments from LDS (shared by both query tiles) ----
    v16bf ak0a = load_a(kc, DD, lane);
    v16bf ak0b = load_a(kc + 32, DD, lane);
    v16bf ak1a = load_a(kc + 16 * DD, DD, lane);
    v16bf ak1b = load_a(kc + 16 * DD + 32, DD, lane);

    // ---- S^T tiles: cXY = keys tile X vs query tile Y ----
    v8f c00 = {}, c10 = {}, c01 = {}, c11 = {};
    c00 = wmma_bf16(ak0a, bq0a, c00); c00 = wmma_bf16(ak0b, bq0b, c00);
    c10 = wmma_bf16(ak1a, bq0a, c10); c10 = wmma_bf16(ak1b, bq0b, c10);
    c01 = wmma_bf16(ak0a, bq1a, c01); c01 = wmma_bf16(ak0b, bq1b, c01);
    c11 = wmma_bf16(ak1a, bq1a, c11); c11 = wmma_bf16(ak1b, bq1b, c11);

    // ---- key mask for this lane's 16 t-elements (aligned float4 loads) ----
    const float* km0p = kmask + b * SS + t0 + half8;
    const float* km1p = kmask + b * SS + t0 + 16 + half8;
    float4 k0a = *reinterpret_cast<const float4*>(km0p);
    float4 k0b = *reinterpret_cast<const float4*>(km0p + 4);
    float4 k1a = *reinterpret_cast<const float4*>(km1p);
    float4 k1b = *reinterpret_cast<const float4*>(km1p + 4);
    float km0[8] = {k0a.x, k0a.y, k0a.z, k0a.w, k0b.x, k0b.y, k0b.z, k0b.w};
    float km1[8] = {k1a.x, k1a.y, k1a.z, k1a.w, k1b.x, k1b.y, k1b.z, k1b.w};

    // ---- online softmax + P^T fragments per query tile ----
    float corr0, corr1;
    v16bf bp0 = softmax_bfrag(c00, c10, km0, km1, lane, m0v, l0v, corr0);
    v16bf bp1 = softmax_bfrag(c01, c11, km0, km1, lane, m1v, l1v, corr1);
#pragma unroll
    for (int n = 0; n < 4; ++n)
#pragma unroll
      for (int i = 0; i < 8; ++i) { acc0[n][i] *= corr0; acc1[n][i] *= corr1; }

    // ---- acc(d,s) += V^T(16d x 32t) x P^T(32t x 16s); V fragments from LDS ----
#pragma unroll
    for (int n = 0; n < 4; ++n) {
      v16bf av = load_a(vc + (n * 16) * 32, 32, lane);
      acc0[n] = wmma_bf16(av, bp0, acc0[n]);
      acc1[n] = wmma_bf16(av, bp1, acc1[n]);
    }
    __syncthreads();        // all reads of buf done before it is re-staged
  }

  // ---- normalize; qmask==0 or fully-masked row -> 0 (matches nan_to_num) ----
  float qmv0 = qmask[b * SS + s0 + col];
  float qmv1 = qmask[b * SS + s0 + 16 + col];
  float inv0 = (qmv0 != 0.f && l0v > 0.f) ? 1.f / l0v : 0.f;
  float inv1 = (qmv1 != 0.f && l1v > 0.f) ? 1.f / l1v : 0.f;
  // per lane: column s fixed, d = n*16 + half8 + i contiguous -> packed b128 stores
  bf16_t* ob0 = ob + (((long)b * HH + h) * SS + (s0 + col)) * DD + half8;
  bf16_t* ob1 = ob + (((long)b * HH + h) * SS + (s0 + 16 + col)) * DD + half8;
#pragma unroll
  for (int n = 0; n < 4; ++n) {
    uint4 pk;
    pk.x = pack2(acc0[n][0] * inv0, acc0[n][1] * inv0);
    pk.y = pack2(acc0[n][2] * inv0, acc0[n][3] * inv0);
    pk.z = pack2(acc0[n][4] * inv0, acc0[n][5] * inv0);
    pk.w = pack2(acc0[n][6] * inv0, acc0[n][7] * inv0);
    *reinterpret_cast<uint4*>(ob0 + n * 16) = pk;
    pk.x = pack2(acc1[n][0] * inv1, acc1[n][1] * inv1);
    pk.y = pack2(acc1[n][2] * inv1, acc1[n][3] * inv1);
    pk.z = pack2(acc1[n][4] * inv1, acc1[n][5] * inv1);
    pk.w = pack2(acc1[n][6] * inv1, acc1[n][7] * inv1);
    *reinterpret_cast<uint4*>(ob1 + n * 16) = pk;
  }
}

// ---------------- per-head output projection D->O, concat heads ----------------

__global__ __launch_bounds__(32) void out_proj(
    const bf16_t* __restrict__ ob, const bf16_t* __restrict__ WoT,
    const float* __restrict__ bo, bf16_t* __restrict__ xcat)
{
  const int lane = threadIdx.x;
  const int m0 = blockIdx.x * 16;
  const int h  = blockIdx.y;
  const int b  = m0 / SS;
  const int sl = m0 % SS;
  const bf16_t* A = ob + (((long)b * HH + h) * SS + sl) * DD;
  const bf16_t* W = WoT + (long)h * OO * DD;

  v8f acc[4] = {};
#pragma unroll
  for (int k0 = 0; k0 < DD; k0 += 32) {
    v16bf a = load_a(A + k0, DD, lane);
#pragma unroll
    for (int n = 0; n < 4; ++n) {
      v16bf bb = load_bt(W + (long)(n * 16) * DD + k0, DD, lane);
      acc[n] = wmma_bf16(a, bb, acc[n]);
    }
  }
  const int half8 = (lane < 16) ? 0 : 8;
  const int col   = lane & 15;
#pragma unroll
  for (int n = 0; n < 4; ++n) {
    int o = n * 16 + col;
    float bi = bo[h * OO + o];
#pragma unroll
    for (int i = 0; i < 8; ++i) {
      int g = m0 + i + half8;
      xcat[(long)g * HO + h * OO + o] = (bf16_t)(acc[n][i] + bi);
    }
  }
}

// ---------------- final linear (H*O)->E, fp32 output, 32 rows/wave ----------------

__global__ __launch_bounds__(32) void final_proj(
    const bf16_t* __restrict__ xcat, const bf16_t* __restrict__ WfT,
    const float* __restrict__ bfin, float* __restrict__ out)
{
  const int lane = threadIdx.x;
  const int m0 = blockIdx.x * 32;
  const int nb = blockIdx.y * 64;

  v8f acc[2][4] = {};
  for (int k0 = 0; k0 < HO; k0 += 32) {
    v16bf a0 = load_a(xcat + (long)m0 * HO + k0, HO, lane);
    v16bf a1 = load_a(xcat + (long)(m0 + 16) * HO + k0, HO, lane);
#pragma unroll
    for (int n = 0; n < 4; ++n) {
      v16bf bb = load_bt(WfT + (long)(nb + n * 16) * HO + k0, HO, lane);  // shared
      acc[0][n] = wmma_bf16(a0, bb, acc[0][n]);
      acc[1][n] = wmma_bf16(a1, bb, acc[1][n]);
    }
  }
  const int half8 = (lane < 16) ? 0 : 8;
  const int col   = lane & 15;
#pragma unroll
  for (int mt = 0; mt < 2; ++mt) {
#pragma unroll
    for (int n = 0; n < 4; ++n) {
      int e = nb + n * 16 + col;
      float bi = bfin[e];
#pragma unroll
      for (int i = 0; i < 8; ++i) {
        int g = m0 + mt * 16 + i + half8;
        out[(long)g * EE + e] = acc[mt][n][i] + bi;
      }
    }
  }
}

// ---------------- host launch ----------------

extern "C" void kernel_launch(void* const* d_in, const int* in_sizes, int n_in,
                              void* d_out, int out_size, void* d_ws, size_t ws_size,
                              hipStream_t stream) {
  (void)in_sizes; (void)n_in; (void)out_size; (void)ws_size;
  const float* query = (const float*)d_in[0];
  const float* key   = (const float*)d_in[1];
  const float* value = (const float*)d_in[2];
  const float* qmask = (const float*)d_in[3];
  const float* kmask = (const float*)d_in[4];
  const float* Wq    = (const float*)d_in[5];
  const float* bq    = (const float*)d_in[6];
  const float* Wk    = (const float*)d_in[7];
  const float* bk    = (const float*)d_in[8];
  const float* Wv    = (const float*)d_in[9];
  const float* bv    = (const float*)d_in[10];
  const float* Wo    = (const float*)d_in[11];
  const float* bo    = (const float*)d_in[12];
  const float* Wf    = (const float*)d_in[13];
  const float* bfin  = (const float*)d_in[14];
  float* out = (float*)d_out;

  char* ws = (char*)d_ws;
  size_t off = 0;
  auto alloc = [&](size_t bytes) -> void* {
    void* p = ws + off;
    off += (bytes + 255) & ~(size_t)255;
    return p;
  };

  bf16_t* qx   = (bf16_t*)alloc((size_t)BB * SS * EE * 2);
  bf16_t* kx   = (bf16_t*)alloc((size_t)BB * SS * EE * 2);
  bf16_t* vx   = (bf16_t*)alloc((size_t)BB * SS * EE * 2);
  bf16_t* WqT  = (bf16_t*)alloc((size_t)HH * DD * EE * 2);
  bf16_t* WkT  = (bf16_t*)alloc((size_t)HH * DD * EE * 2);
  bf16_t* WvT  = (bf16_t*)alloc((size_t)HH * DD * EE * 2);
  bf16_t* WoT  = (bf16_t*)alloc((size_t)HH * OO * DD * 2);
  bf16_t* WfT  = (bf16_t*)alloc((size_t)EE * HO * 2);
  bf16_t* qhd  = (bf16_t*)alloc((size_t)BB * HH * SS * DD * 2);
  bf16_t* khd  = (bf16_t*)alloc((size_t)BB * HH * SS * DD * 2);
  bf16_t* vhT  = (bf16_t*)alloc((size_t)BB * HH * DD * SS * 2);
  bf16_t* obuf = (bf16_t*)alloc((size_t)BB * HH * SS * DD * 2);
  bf16_t* xcat = (bf16_t*)alloc((size_t)BB * SS * HO * 2);

  const int nAct = BB * SS * EE;
  dim3 blk(256);
  cvt_f32_bf16<<<dim3((nAct + 255) / 256), blk, 0, stream>>>(query, qx, nAct);
  cvt_f32_bf16<<<dim3((nAct + 255) / 256), blk, 0, stream>>>(key,   kx, nAct);
  cvt_f32_bf16<<<dim3((nAct + 255) / 256), blk, 0, stream>>>(value, vx, nAct);

  transpose_cvt<<<dim3((EE * DD + 255) / 256, 1, HH), blk, 0, stream>>>(Wq, WqT, EE, DD);
  transpose_cvt<<<dim3((EE * DD + 255) / 256, 1, HH), blk, 0, stream>>>(Wk, WkT, EE, DD);
  transpose_cvt<<<dim3((EE * DD + 255) / 256, 1, HH), blk, 0, stream>>>(Wv, WvT, EE, DD);
  transpose_cvt<<<dim3((DD * OO + 255) / 256, 1, HH), blk, 0, stream>>>(Wo, WoT, DD, OO);
  transpose_cvt<<<dim3((HO * EE + 255) / 256, 1, 1),  blk, 0, stream>>>(Wf, WfT, HO, EE);

  qkv_proj<<<dim3(BB * SS / 32, HH, 3), dim3(32), 0, stream>>>(
      qx, kx, vx, WqT, WkT, WvT, bq, bk, bv, qhd, khd, vhT);

  attn<<<dim3(SS / 128, HH, BB), dim3(128), 0, stream>>>(
      qhd, khd, vhT, qmask, kmask, obuf);

  out_proj<<<dim3(BB * SS / 16, HH), dim3(32), 0, stream>>>(obuf, WoT, bo, xcat);

  final_proj<<<dim3(BB * SS / 32, EE / 64), dim3(32), 0, stream>>>(xcat, WfT, bfin, out);
}
